// Gru4RecModel_53721450938923
// MI455X (gfx1250) — matrix operand order repaired and checked by reading, not compiled, not run on hardware
//
#include <hip/hip_runtime.h>
#include <hip/hip_bf16.h>
#include <math.h>

// Problem sizes (fixed by the reference)
#define VSZ 100001   // V + 1
#define H   128
#define BB  1024
#define TT  50
#define H3  384

// ---------------------------------------------------------------------------
// WMMA / vector types (CDNA5 / gfx1250, wave32)
// ---------------------------------------------------------------------------
typedef __attribute__((ext_vector_type(16))) __bf16 bf16x16;
typedef __attribute__((ext_vector_type(8)))  float  f32x8;
typedef __attribute__((ext_vector_type(4)))  unsigned int u32x4;
typedef __attribute__((ext_vector_type(8)))  int    i32x8;
typedef __attribute__((ext_vector_type(4)))  int    i32x4;

union Frag {
    bf16x16        v;
    unsigned short h[16];
    unsigned int   w[8];
    u32x4          q[2];
};

__device__ __forceinline__ unsigned short f2bf(float f) {
    union { float f; unsigned u; } c; c.f = f;
    unsigned r = c.u + 0x7FFFu + ((c.u >> 16) & 1u);   // round-to-nearest-even
    return (unsigned short)(r >> 16);
}

__device__ __forceinline__ f32x8 wmma_bf16(bf16x16 a, bf16x16 b, f32x8 c) {
    // 8 args: (neg_a, A, neg_b, B, c_mod, C, reuse_a, reuse_b)
    return __builtin_amdgcn_wmma_f32_16x16x32_bf16(false, a, false, b,
                                                   (short)0, c, false, false);
}

// A-fragment element -> K offset within a 16x32 tile (ISA 7.12.2):
// lanes 0-15:  K = e + (e>=8 ? 8 : 0)       (K 0-7, 16-23)
// lanes 16-31: K = 8 + e + (e>=8 ? 8 : 0)   (K 8-15, 24-31)
#define A_K(kb, e) ((kb) + (e) + (((e) >= 8) ? 8 : 0))

// ---------------------------------------------------------------------------
// Kernel 1: fused embedding gather + input projection
//   xp[b*T+t, 0:3H] = emb[ids[b,t]] @ gru_kernel + b_i
// block = 256 thr (8 waves); wave -> 16 rows; block -> 128 rows; grid = 400
// gru_kernel staged once per block in LDS, bf16, N-major [384][128] (96 KB).
// ---------------------------------------------------------------------------
__global__ void k_embed_proj(const int* __restrict__ ids,
                             const float* __restrict__ emb,
                             const float* __restrict__ kern,
                             const float* __restrict__ bias_i,
                             float* __restrict__ xp) {
    extern __shared__ unsigned short lk[];            // [H3][H] bf16 N-major
    const int tid = threadIdx.x;

    for (int idx = tid; idx < H3 * H; idx += 256) {
        int n = idx >> 7, k = idx & 127;
        lk[idx] = f2bf(kern[k * H3 + n]);             // transpose to N-major
    }
    __syncthreads();

    const int lane = tid & 31, wave = tid >> 5;
    const int m0   = blockIdx.x * 128 + wave * 16;
    const int r    = lane & 15;
    const int kbA  = (lane < 16) ? 0 : 16;
    const int n    = lane & 15;
    const int kbB  = (lane < 16) ? 0 : 16;

    // Gather embedding row for this lane's M-row, convert to bf16 A-frags
    const float* ep = emb + (long long)ids[m0 + r] * H;
    Frag a[4];
#pragma unroll
    for (int c = 0; c < 4; ++c)
#pragma unroll
        for (int e = 0; e < 16; ++e)
            a[c].h[e] = f2bf(ep[32 * c + kbA + e]);

    for (int nt = 0; nt < 24; ++nt) {                 // 24 * 16 = 384 cols
        const int n0 = nt * 16;
        f32x8 acc = {};
#pragma unroll
        for (int c = 0; c < 4; ++c) {
            Frag b;
            const unsigned int* p =
                (const unsigned int*)(lk + (n0 + n) * H + 32 * c + kbB);
#pragma unroll
            for (int wd = 0; wd < 8; ++wd) b.w[wd] = p[wd];
            acc = wmma_bf16(a[c].v, b.v, acc);
        }
        const float bi = bias_i[n0 + n];
#pragma unroll
        for (int rr = 0; rr < 8; ++rr) {
            int m = m0 + ((lane < 16) ? rr : 8 + rr);
            xp[(long long)m * H3 + n0 + n] = acc[rr] + bi;
        }
    }
}

// ---------------------------------------------------------------------------
// Kernel 2: sequential GRU scan (reset_after=True), 16 batch rows per block.
// rec_kernel resident in LDS (bf16 N-major, 96 KB) for all 50 steps;
// h kept in LDS (fp32). grid = 64 blocks, 128 thr (4 waves).
// Captures h at t == len-1, written DIRECTLY in pre-swizzled WMMA A-fragment
// order so kernel 3 can load A chunks with contiguous b128s:
//   lastA[(((mt*4 + c)*32 + lane) << 4) + e] , mt = row/16
// ---------------------------------------------------------------------------
__global__ void k_gru_scan(const float* __restrict__ xp,
                           const float* __restrict__ rkern,
                           const float* __restrict__ bias,   // [2][3H]
                           const int* __restrict__ mask,
                           unsigned short* __restrict__ lastA) {
    extern __shared__ unsigned char smem[];
    unsigned short* lk   = (unsigned short*)smem;               // [H3][H] bf16
    float*          hbuf = (float*)(smem + (size_t)H3 * H * 2); // [16][H]
    float*          rec  = hbuf + 16 * H;                       // [16][H3]
    int*            len  = (int*)(rec + 16 * H3);               // [16]

    const int tid = threadIdx.x;                                 // 128 threads
    for (int idx = tid; idx < H3 * H; idx += 128) {
        int nn = idx >> 7, k = idx & 127;
        lk[idx] = f2bf(rkern[k * H3 + nn]);
    }
    for (int idx = tid; idx < 16 * H; idx += 128) hbuf[idx] = 0.f;

    const int b0 = blockIdx.x * 16;
    if (tid < 16) {
        int s = 0;
        for (int t = 0; t < TT; ++t) s += mask[(b0 + tid) * TT + t];
        len[tid] = s;
    }
    __syncthreads();

    const int lane = tid & 31, wave = tid >> 5;
    const int r    = lane & 15;
    const int kbA  = (lane < 16) ? 0 : 16;
    const int n    = lane & 15;
    const int kbB  = (lane < 16) ? 0 : 16;
    const float* b_r = bias + H3;

    // Swizzle for this thread's hidden column (tid = k in 0..127)
    const int ck  = tid >> 5;         // K chunk
    const int lk2 = tid & 31;         // K within chunk
    int laneS, eS;
    if      (lk2 <  8) { laneS = 0;  eS = lk2;      }
    else if (lk2 < 16) { laneS = 16; eS = lk2 - 8;  }
    else if (lk2 < 24) { laneS = 0;  eS = lk2 - 8;  }
    else               { laneS = 16; eS = lk2 - 16; }

    for (int t = 0; t < TT; ++t) {
        // A-frags from current hidden state (fp32 LDS -> bf16)
        Frag a[4];
#pragma unroll
        for (int c = 0; c < 4; ++c)
#pragma unroll
            for (int e = 0; e < 16; ++e)
                a[c].h[e] = f2bf(hbuf[r * H + 32 * c + kbA + e]);

        // rec = h @ rec_kernel + b_r ; wave handles 6 N-tiles (96 cols)
#pragma unroll
        for (int j = 0; j < 6; ++j) {
            const int n0 = (wave * 6 + j) * 16;
            f32x8 acc = {};
#pragma unroll
            for (int c = 0; c < 4; ++c) {
                Frag b;
                const unsigned int* p =
                    (const unsigned int*)(lk + (n0 + n) * H + 32 * c + kbB);
#pragma unroll
                for (int wd = 0; wd < 8; ++wd) b.w[wd] = p[wd];
                acc = wmma_bf16(a[c].v, b.v, acc);
            }
            const float br = b_r[n0 + n];
#pragma unroll
            for (int rr = 0; rr < 8; ++rr) {
                int m = (lane < 16) ? rr : 8 + rr;
                rec[m * H3 + n0 + n] = acc[rr] + br;
            }
        }
        __syncthreads();

        // Gate math: thread tid owns hidden column tid for all 16 rows
#pragma unroll 4
        for (int m = 0; m < 16; ++m) {
            const long long xrow = (long long)(b0 + m) * TT + t;
            float xz = xp[xrow * H3 + tid];
            float xr = xp[xrow * H3 + H + tid];
            float xh = xp[xrow * H3 + 2 * H + tid];
            float rz = rec[m * H3 + tid];
            float rr_ = rec[m * H3 + H + tid];
            float rh = rec[m * H3 + 2 * H + tid];
            float z  = 1.f / (1.f + __expf(-(xz + rz)));
            float rg = 1.f / (1.f + __expf(-(xr + rr_)));
            float hh = tanhf(xh + rg * rh);
            float hp = hbuf[m * H + tid];
            float hn = z * hp + (1.f - z) * hh;
            hbuf[m * H + tid] = hn;
            if (t == len[m] - 1) {
                const int mt = (b0 + m) >> 4;
                lastA[(((mt * 4 + ck) * 32 + laneS + m) << 4) + eS] = f2bf(hn);
            }
        }
        __syncthreads();
    }
}

// ---------------------------------------------------------------------------
// Kernel 3: logits = last @ emb^T  ([1024x128] x [128x100001])
// Write-bandwidth bound (410 MB out). Block = 256 thr (8 waves) owns one
// 16-column vocab tile. The 16 emb rows (16x128 fp32, stride 128) are DMA'd
// into LDS by the Tensor Data Mover (tensor_load_to_lds + s_wait_tensorcnt);
// TDM zero-fills rows past tensor_dim1, handling the vocab tail for free.
// A-operand comes pre-swizzled from kernel 2 -> contiguous b128 loads.
// ---------------------------------------------------------------------------
__global__ void k_logits(const unsigned short* __restrict__ lastA,
                         const float* __restrict__ emb,
                         float* __restrict__ out) {
    __shared__ float ebf[16 * H];                     // 8 KB at LDS offset 0
    const int tid = threadIdx.x;
    const int n0  = blockIdx.x * 16;

    if (tid < 32) {                                   // wave 0 issues the TDM op
        unsigned long long ga =
            (unsigned long long)(const void*)(emb + (unsigned long long)n0 * H);
        unsigned rem = (unsigned)(VSZ - n0);          // rows left in vocab
        // D# group 0: count=1 | lds_addr=0 | global_addr | type=2
        u32x4 g0 = { 1u,
                     0u,
                     (unsigned)(ga & 0xFFFFFFFFull),
                     (unsigned)((ga >> 32) & 0x1FFFFFFull) | (2u << 30) };
        // D# group 1: data_size=4B, tensor_dim0=128, tensor_dim1=rem,
        //             tile_dim0=128, tile_dim1=16, tensor_dim0_stride=128
        i32x8 g1 = { (int)(2u << 16),                 // data_size = 4 bytes
                     (int)(128u << 16),               // tensor_dim0[15:0]
                     (int)((rem & 0xFFFFu) << 16),    // tensor_dim1[15:0]
                     (int)((rem >> 16) | (128u << 16)), // dim1 hi | tile_dim0
                     16,                              // tile_dim1 = 16 rows
                     128,                             // tensor_dim0_stride
                     0, 0 };
        i32x4 gz4 = { 0, 0, 0, 0 };
        i32x8 gz8 = { 0, 0, 0, 0, 0, 0, 0, 0 };
        // 6-arg toolchain variant: (g0, g1, g2, g3, g4, cpol)
        __builtin_amdgcn_tensor_load_to_lds(g0, g1, gz4, gz4, gz8, 0);
        __builtin_amdgcn_s_wait_tensorcnt(0);
    }
    __syncthreads();

    const int lane = tid & 31, wave = tid >> 5;
    const int n    = lane & 15;
    const int kbB  = (lane < 16) ? 0 : 8;

    Frag bfrag[4];                                    // B held for whole block
#pragma unroll
    for (int c = 0; c < 4; ++c)
#pragma unroll
        for (int e = 0; e < 16; ++e)
            bfrag[c].h[e] = f2bf(ebf[n * H + 32 * c + A_K(kbB, e)]);

    const int  col = n0 + n;
    const bool cok = (col < VSZ);

    for (int mt = wave; mt < 64; mt += 8) {           // 64 M-tiles of 16 rows
        if (mt + 8 < 64)                              // global_prefetch_b8
            __builtin_prefetch(lastA + ((((mt + 8) * 4) * 32 + lane) << 4), 0, 0);
        f32x8 acc = {};
#pragma unroll
        for (int c = 0; c < 4; ++c) {
            Frag a;
            const u32x4* p =
                (const u32x4*)(lastA + (((mt * 4 + c) * 32 + lane) << 4));
            a.q[0] = p[0];
            a.q[1] = p[1];
            acc = wmma_bf16(a.v, bfrag[c].v, acc);
        }
        if (cok) {
#pragma unroll
            for (int rr = 0; rr < 8; ++rr) {
                int m = mt * 16 + ((lane < 16) ? rr : 8 + rr);
                out[(long long)m * VSZ + col] = acc[rr];
            }
        }
    }
}

// ---------------------------------------------------------------------------
// Launch
// ---------------------------------------------------------------------------
extern "C" void kernel_launch(void* const* d_in, const int* in_sizes, int n_in,
                              void* d_out, int out_size, void* d_ws, size_t ws_size,
                              hipStream_t stream) {
    const int*   ids   = (const int*)d_in[0];   // input_ids   [B,T]
    const int*   mask  = (const int*)d_in[1];   // input_mask  [B,T]
    const float* emb   = (const float*)d_in[2]; // emb_table   [V+1,H]
    const float* kern  = (const float*)d_in[3]; // gru_kernel  [H,3H]
    const float* rkern = (const float*)d_in[4]; // gru_rec_kernel [H,3H]
    const float* bias  = (const float*)d_in[5]; // gru_bias    [2,3H]
    float* out = (float*)d_out;

    char* ws = (char*)d_ws;
    float* xp = (float*)ws;                                 // 51200*384 f32
    size_t off = (size_t)BB * TT * H3 * sizeof(float);      // 78.6 MB
    unsigned short* lastA = (unsigned short*)(ws + off);    // 1024*128 bf16 (swizzled)

    // 1) embed + input projection  (96 KB dynamic LDS for bf16 weights)
    hipLaunchKernelGGL(k_embed_proj, dim3((BB * TT) / 128), dim3(256),
                       (size_t)H3 * H * 2, stream, ids, emb, kern, bias, xp);

    // 2) GRU scan  (96 KB weights + 8 KB h + 24 KB rec + lens in LDS)
    size_t smem2 = (size_t)H3 * H * 2 + 16 * H * 4 + 16 * H3 * 4 + 64;
    hipLaunchKernelGGL(k_gru_scan, dim3(BB / 16), dim3(128), smem2, stream,
                       xp, rkern, bias, mask, lastA);

    // 3) tied-embedding logits (TDM-staged B operand)
    hipLaunchKernelGGL(k_logits, dim3((VSZ + 15) / 16), dim3(256), 0, stream,
                       lastA, emb, out);
}
